// LinearAttention_1408749273192
// MI455X (gfx1250) — matrix operand order repaired and checked by reading, compiled, not run on hardware
//
#include <hip/hip_runtime.h>

// ---------------------------------------------------------------------------
// Linear attention for MI455X (gfx1250).
//   * all GEMMs: v_wmma_f32_16x16x32_bf16 (wave32), f32 accumulate
//   * A-operands staged global->LDS by the Tensor Data Mover
//     (tensor_load_to_lds, padded rows via D# pad fields, s_wait_tensorcnt)
//   * B-operands: fragment-major swizzled weights read straight from L2
//     with 2x global_load_b128 per fragment
//   * kv_sum: k-pair-packed LDS staging -> ds_load_2addr_b32 fragments
// ---------------------------------------------------------------------------

#define DEV __device__ __forceinline__

typedef __attribute__((ext_vector_type(16))) __bf16 v16bf;
typedef __attribute__((ext_vector_type(8)))  float  v8f;
typedef unsigned int v4u __attribute__((ext_vector_type(4)));
typedef int          v8i __attribute__((ext_vector_type(8)));
typedef int          v4i __attribute__((ext_vector_type(4)));

union FragU {
  v16bf v;
  uint4 q[2];
  unsigned int u[8];
  unsigned short s[16];
};

static constexpr int D_MODEL   = 1024;
static constexpr int NUM_HEADS = 16;
static constexpr int HEAD_DIM  = 64;
static constexpr int BATCH     = 4;
static constexpr int T_SEQ     = 4096;
static constexpr int M_ROWS    = BATCH * T_SEQ;     // 16384
static constexpr int BH        = BATCH * NUM_HEADS; // 64
static constexpr int KSPLIT    = 8;

DEV unsigned short f2bf(float f) {
  unsigned int u = __float_as_uint(f);
  u += 0x7FFFu + ((u >> 16) & 1u);  // RNE
  return (unsigned short)(u >> 16);
}
DEV float bf2f(unsigned short h) {
  return __uint_as_float((unsigned int)h << 16);
}

DEV v8f wmma_bf16(v16bf a, v16bf b, v8f c) {
  return __builtin_amdgcn_wmma_f32_16x16x32_bf16(
      false, a, false, b, (short)0, c, false, false);
}

// ---------------------------------------------------------------------------
// TDM: 2D bf16 tile global->LDS with row padding.
//   tile_w elems/row, tile_h rows, row_stride elems between global rows,
//   pad_code: pad after 2^(pad_code+1) DWORDs, pad_dw: pad amount in DWORDs.
// Issue from ONE wave only (TDM ignores EXEC). Wait with s_wait_tensorcnt.
// ---------------------------------------------------------------------------
DEV void tdm_load_2d_bf16(unsigned int lds_byte_addr, const void* gptr,
                          int tile_w, int tile_h, int row_stride,
                          int pad_code, int pad_dw) {
  const unsigned long long ga = (unsigned long long)(size_t)gptr;
  v4u g0;
  g0[0] = 1u;  // count=1, user-mode descriptor
  g0[1] = lds_byte_addr;
  g0[2] = (unsigned int)(ga & 0xffffffffu);
  g0[3] = (unsigned int)((ga >> 32) & 0x01ffffffu) | (2u << 30);  // type=2 (image)
  const unsigned int td0 = 1u << 30;  // tensor dims large: never OOB
  const unsigned int td1 = 1u << 20;
  v8i g1;
  g1[0] = (int)((1u << 16) |                 // data_size = 2 bytes
                (1u << 20) |                 // pad_enable
                ((unsigned)pad_code << 22) |
                ((unsigned)(pad_dw - 1) << 25));
  g1[1] = (int)((td0 & 0xffffu) << 16);                       // dim0 lo16 (atomic addr 0)
  g1[2] = (int)((td0 >> 16) | ((td1 & 0xffffu) << 16));       // dim0 hi16 | dim1 lo16
  g1[3] = (int)((td1 >> 16) | ((unsigned)tile_w << 16));      // dim1 hi16 | tile_dim0
  g1[4] = (int)((unsigned)tile_h);                            // tile_dim1 | tile_dim2=0
  g1[5] = (int)((unsigned)row_stride);                        // dim0_stride low32
  g1[6] = 0;
  g1[7] = 0;
  const v4i z4 = {0, 0, 0, 0};
#if defined(__clang_major__) && (__clang_major__ >= 23)
  const v8i z8 = {0, 0, 0, 0, 0, 0, 0, 0};
  __builtin_amdgcn_tensor_load_to_lds(g0, g1, z4, z4, z8, 0);
#else
  __builtin_amdgcn_tensor_load_to_lds(g0, g1, z4, z4, 0);
#endif
}

// A fragment (16x32 bf16) from row-major LDS tile; ld multiple of 8 ushorts.
DEV v16bf load_frag_a_b128(const unsigned short* base, int ld, int m0, int lane) {
  const uint4* rp = (const uint4*)(base + (size_t)(m0 + (lane & 15)) * ld);
  const int h8 = lane >> 4;
  FragU f;
  f.q[0] = rp[2 * h8];
  f.q[1] = rp[2 * h8 + 1];
  return f.v;
}

// B fragment (32x16 bf16) from fragment-major swizzled global memory.
DEV v16bf load_frag_b_swz(const unsigned int* Wswz, int frag_id, int lane) {
  const uint4* p = (const uint4*)(Wswz + ((size_t)frag_id * 32 + lane) * 8);
  FragU f;
  f.q[0] = p[0];
  f.q[1] = p[1];
  return f.v;
}

// A fragment, logical A = transpose of k-pair-packed LDS tile.
DEV v16bf load_frag_a_pairpacked(const unsigned int* base, int ldp, int m0, int lane) {
  const int m = m0 + (lane & 15);
  const int half = lane >> 4;
  FragU f;
#pragma unroll
  for (int j = 0; j < 8; ++j) {
    const int kp = 8 * half + j;
    f.u[j] = base[kp * ldp + m];
  }
  return f.v;
}

// B fragment from k-pair-packed LDS tile.
DEV v16bf load_frag_b_pairpacked(const unsigned int* base, int ldp, int n0, int lane) {
  const int n = n0 + (lane & 15);
  const int kg = lane >> 4;
  FragU f;
#pragma unroll
  for (int v = 0; v < 8; ++v) f.u[v] = base[(kg * 8 + v) * ldp + n];
  return f.v;
}

// ---------------------------------------------------------------------------
// fp32 -> bf16 row-major conversion (8 floats per thread)
// ---------------------------------------------------------------------------
__global__ __launch_bounds__(256) void convert_f32_bf16(const float* __restrict__ src,
                                                        unsigned int* __restrict__ dst) {
  const int i = blockIdx.x * 256 + threadIdx.x;
  const float4 a = ((const float4*)src)[2 * i];
  const float4 b = ((const float4*)src)[2 * i + 1];
  uint4 p;
  p.x = (unsigned int)f2bf(a.x) | ((unsigned int)f2bf(a.y) << 16);
  p.y = (unsigned int)f2bf(a.z) | ((unsigned int)f2bf(a.w) << 16);
  p.z = (unsigned int)f2bf(b.x) | ((unsigned int)f2bf(b.y) << 16);
  p.w = (unsigned int)f2bf(b.z) | ((unsigned int)f2bf(b.w) << 16);
  ((uint4*)dst)[i] = p;
}

// ---------------------------------------------------------------------------
// Weight prep into fragment-major swizzled bf16 uints.
// ---------------------------------------------------------------------------
__global__ void prep_qkv_weights(const float* __restrict__ Wq,
                                 const float* __restrict__ Wk,
                                 const float* __restrict__ Wv,
                                 unsigned int* __restrict__ oq,
                                 unsigned int* __restrict__ ok,
                                 unsigned int* __restrict__ ov) {
  const int idx = blockIdx.x * 256 + threadIdx.x;  // D*D/2 uints
  const int v = idx & 7;
  const int lane = (idx >> 3) & 31;
  const int nt = (idx >> 8) & 63;
  const int kt = idx >> 14;
  const int k = kt * 32 + (lane >> 4) * 16 + 2 * v;
  const int n = nt * 16 + (lane & 15);
  const int h = n >> 6, e = n & 63;
  const size_t s0 = ((size_t)h * D_MODEL + k) * HEAD_DIM + e;
  const size_t s1 = s0 + HEAD_DIM;
  oq[idx] = (unsigned int)f2bf(Wq[s0]) | ((unsigned int)f2bf(Wq[s1]) << 16);
  ok[idx] = (unsigned int)f2bf(Wk[s0]) | ((unsigned int)f2bf(Wk[s1]) << 16);
  ov[idx] = (unsigned int)f2bf(Wv[s0]) | ((unsigned int)f2bf(Wv[s1]) << 16);
}

__global__ void prep_wo(const float* __restrict__ Wo, unsigned int* __restrict__ oT) {
  const int idx = blockIdx.x * 256 + threadIdx.x;
  const int v = idx & 7;
  const int lane = (idx >> 3) & 31;
  const int nt = (idx >> 8) & 63;
  const int kt = idx >> 14;
  const int k = kt * 32 + (lane >> 4) * 16 + 2 * v;
  const int n = nt * 16 + (lane & 15);
  const float a = Wo[(size_t)n * D_MODEL + k];
  const float b = Wo[(size_t)n * D_MODEL + k + 1];
  oT[idx] = (unsigned int)f2bf(a) | ((unsigned int)f2bf(b) << 16);
}

// ---------------------------------------------------------------------------
// 128x128x32 bf16 WMMA GEMM; A staged by TDM, B direct from swizzled global.
// ---------------------------------------------------------------------------
template <bool SQUARE, bool BF16OUT>
__global__ __launch_bounds__(256) void wmma_gemm_kernel(
    const unsigned short* __restrict__ X, const unsigned int* __restrict__ Wswz,
    const float* __restrict__ bias, void* __restrict__ Yv,
    int M, int K, int N) {
  constexpr int BM = 128, BN = 128, BK = 32;
  constexpr int LDA = 40;  // ushorts: 32 + 4DW pad from TDM
  __shared__ unsigned short sA[2][BM * LDA];
  __shared__ float sBias[BN];

  const int tid = threadIdx.x, lane = tid & 31, w = tid >> 5;
  const int row0 = blockIdx.x * BM, col0 = blockIdx.y * BN;
  if (tid < BN) sBias[tid] = bias[col0 + tid];

  const int KT = K / BK, NT = N >> 4;
  const int wm = w >> 2, wn = w & 3;
  const int ntg0 = (col0 >> 4) + wn * 2;
  v8f acc[4][2] = {};

  // TDM: tile 32x128, global row stride K elems, rows 16DW + 4DW pad -> LDA=40
  if (w == 0) {
    tdm_load_2d_bf16((unsigned int)(size_t)&sA[0][0], X + (size_t)row0 * K,
                     BK, BM, K, /*pad_code=*/3, /*pad_dw=*/4);
    __builtin_amdgcn_s_wait_tensorcnt(0);
  }
  __syncthreads();

  for (int kt = 0; kt < KT; ++kt) {
    const int buf = kt & 1;
    if (w == 0 && kt + 1 < KT)
      tdm_load_2d_bf16((unsigned int)(size_t)&sA[buf ^ 1][0],
                       X + (size_t)row0 * K + (kt + 1) * BK, BK, BM, K, 3, 4);
    v16bf bfr[2];
#pragma unroll
    for (int j = 0; j < 2; ++j)
      bfr[j] = load_frag_b_swz(Wswz, kt * NT + ntg0 + j, lane);
#pragma unroll
    for (int i = 0; i < 4; ++i) {
      v16bf afr = load_frag_a_b128(&sA[buf][0], LDA, wm * 64 + i * 16, lane);
#pragma unroll
      for (int j = 0; j < 2; ++j) acc[i][j] = wmma_bf16(afr, bfr[j], acc[i][j]);
    }
    if (w == 0) __builtin_amdgcn_s_wait_tensorcnt(0);
    __syncthreads();
  }

  const int half = lane >> 4, nn = lane & 15;
#pragma unroll
  for (int i = 0; i < 4; ++i) {
#pragma unroll
    for (int j = 0; j < 2; ++j) {
#pragma unroll
      for (int e = 0; e < 8; ++e) {
        const int r = row0 + wm * 64 + i * 16 + e + 8 * half;
        const int c = col0 + wn * 32 + j * 16 + nn;
        float val = acc[i][j][e] + sBias[wn * 32 + j * 16 + nn];
        if constexpr (SQUARE) val = val * val;
        if constexpr (BF16OUT) {
          const int b = r >> 12;
          const int t = r & (T_SEQ - 1);
          const int h = c >> 6;
          const int e2 = c & 63;
          ((unsigned short*)Yv)[((((size_t)b * NUM_HEADS + h) * T_SEQ) + t) * HEAD_DIM + e2] =
              f2bf(val);
        } else {
          ((float*)Yv)[(size_t)r * N + c] = val;
        }
      }
    }
  }
}

// ---------------------------------------------------------------------------
// kv_sum split-K partials with k-pair-packed staging.
// ---------------------------------------------------------------------------
__global__ __launch_bounds__(256) void kvsum_partial_kernel(
    const unsigned short* __restrict__ KH2, const unsigned short* __restrict__ VH,
    float* __restrict__ kv_part, float* __restrict__ ks_part) {
  constexpr int E = HEAD_DIM, TS = T_SEQ / KSPLIT, BK = 32;
  constexpr int LDP = 68;
  __shared__ unsigned int sKp[2][16 * LDP];
  __shared__ unsigned int sVp[2][16 * LDP];
  __shared__ float sred[4][E];

  const int tid = threadIdx.x, lane = tid & 31, w = tid >> 5;
  const int bh = blockIdx.x, split = blockIdx.y;
  const size_t base = ((size_t)bh * T_SEQ + split * TS) * E;

  auto load_chunk = [&](int kc, int buf) {
    const int isV = tid >> 7;
    const int ti = tid & 127;
    const int kp = ti >> 3;
    const int db = (ti & 7) * 8;
    const unsigned short* src = (isV ? VH : KH2) + base + (size_t)(kc * BK + 2 * kp) * E + db;
    uint4 a = *(const uint4*)(src);
    uint4 b = *(const uint4*)(src + E);
    uint4 p0, p1;
    p0.x = (a.x & 0xffffu) | (b.x << 16);
    p0.y = (a.x >> 16) | (b.x & 0xffff0000u);
    p0.z = (a.y & 0xffffu) | (b.y << 16);
    p0.w = (a.y >> 16) | (b.y & 0xffff0000u);
    p1.x = (a.z & 0xffffu) | (b.z << 16);
    p1.y = (a.z >> 16) | (b.z & 0xffff0000u);
    p1.z = (a.w & 0xffffu) | (b.w << 16);
    p1.w = (a.w >> 16) | (b.w & 0xffff0000u);
    uint4* dst = (uint4*)((isV ? &sVp[buf][0] : &sKp[buf][0]) + kp * LDP + db);
    dst[0] = p0;
    dst[1] = p1;
  };

  const int it = w >> 1;
  const int j0 = (w & 1) * 2;
  v8f acc[2] = {};

  load_chunk(0, 0);
  __syncthreads();
  for (int kc = 0; kc < TS / BK; ++kc) {
    const int buf = kc & 1;
    if (kc + 1 < TS / BK) load_chunk(kc + 1, buf ^ 1);
    v16bf afr = load_frag_a_pairpacked(&sKp[buf][0], LDP, it * 16, lane);
#pragma unroll
    for (int jj = 0; jj < 2; ++jj) {
      v16bf bfr = load_frag_b_pairpacked(&sVp[buf][0], LDP, (j0 + jj) * 16, lane);
      acc[jj] = wmma_bf16(afr, bfr, acc[jj]);
    }
    __syncthreads();
  }

  const int half = lane >> 4, nn = lane & 15;
  float* outp = kv_part + ((size_t)bh * KSPLIT + split) * (E * E);
#pragma unroll
  for (int jj = 0; jj < 2; ++jj)
#pragma unroll
    for (int e = 0; e < 8; ++e) {
      const int m = it * 16 + e + 8 * half;
      const int n = (j0 + jj) * 16 + nn;
      outp[m * E + n] = acc[jj][e];
    }

  {
    const int d = tid & 63, g = tid >> 6;
    float s = 0.f;
    const unsigned short* p = KH2 + base + (size_t)g * 128 * E + d;
    for (int t = 0; t < 128; ++t) s += bf2f(p[(size_t)t * E]);
    sred[g][d] = s;
    __syncthreads();
    if (tid < E) {
      const float tot = sred[0][tid] + sred[1][tid] + sred[2][tid] + sred[3][tid];
      ks_part[((size_t)bh * KSPLIT + split) * E + tid] = tot;
    }
  }
}

__global__ __launch_bounds__(256) void kvsum_reduce_kernel(
    const float* __restrict__ kv_part, const float* __restrict__ ks_part,
    unsigned int* __restrict__ kvs_swz, float* __restrict__ ksum_f) {
  const int bh = blockIdx.x, tid = threadIdx.x;
  const float* src = kv_part + (size_t)bh * KSPLIT * 4096;
#pragma unroll
  for (int l = 0; l < 8; ++l) {
    const int idx = tid * 8 + l;
    const int v = idx & 7;
    const int lane = (idx >> 3) & 31;
    const int nt = (idx >> 8) & 3;
    const int kt = idx >> 10;
    const int k = kt * 32 + (lane >> 4) * 16 + 2 * v;
    const int n = nt * 16 + (lane & 15);
    float s0 = 0.f, s1 = 0.f;
#pragma unroll
    for (int sp = 0; sp < KSPLIT; ++sp) {
      s0 += src[(size_t)sp * 4096 + k * HEAD_DIM + n];
      s1 += src[(size_t)sp * 4096 + (k + 1) * HEAD_DIM + n];
    }
    kvs_swz[(size_t)bh * 2048 + idx] =
        (unsigned int)f2bf(s0) | ((unsigned int)f2bf(s1) << 16);
  }
  if (tid < HEAD_DIM) {
    float s = 0.f;
#pragma unroll
    for (int sp = 0; sp < KSPLIT; ++sp)
      s += ks_part[((size_t)bh * KSPLIT + sp) * HEAD_DIM + tid];
    ksum_f[(size_t)bh * HEAD_DIM + tid] = s;
  }
}

// ---------------------------------------------------------------------------
// attn = (qh2 @ kv_sum) * z ; q-tile staged by TDM.
// ---------------------------------------------------------------------------
__global__ __launch_bounds__(256) void attn_kernel(
    const unsigned short* __restrict__ QH2, const unsigned int* __restrict__ KVSW,
    const float* __restrict__ KSUM, unsigned short* __restrict__ ATTN) {
  constexpr int E = HEAD_DIM, BM = 128, LDQ = 72;  // 64 + 4DW pad
  __shared__ unsigned short sQ[BM * LDQ];
  __shared__ float sZ[BM];
  __shared__ float sKS[E];

  const int tid = threadIdx.x, lane = tid & 31, w = tid >> 5;
  const int bh = blockIdx.x, b = bh >> 4, h = bh & 15;
  const int t0 = blockIdx.y * BM;

  // TDM: tile 64x128 bf16, rows 32DW + 4DW pad -> LDQ=72
  if (w == 0) {
    tdm_load_2d_bf16((unsigned int)(size_t)&sQ[0],
                     QH2 + ((size_t)bh * T_SEQ + t0) * E,
                     E, BM, E, /*pad_code=*/4, /*pad_dw=*/4);
  }
  if (tid < E) sKS[tid] = KSUM[(size_t)bh * E + tid];
  if (w == 0) __builtin_amdgcn_s_wait_tensorcnt(0);
  __syncthreads();

  if (tid < BM) {
    const unsigned int* qr = (const unsigned int*)(sQ + tid * LDQ);
    float s = 0.f;
    for (int d2 = 0; d2 < 32; ++d2) {
      const unsigned int u = qr[d2];
      s += bf2f((unsigned short)u) * sKS[2 * d2] +
           bf2f((unsigned short)(u >> 16)) * sKS[2 * d2 + 1];
    }
    sZ[tid] = 1.0f / (s + 1e-6f);
  }
  __syncthreads();

  const int wm = w >> 2, wn = w & 3;
  const unsigned int* kvb = KVSW + (size_t)bh * 2048;
  v8f acc[4] = {};
#pragma unroll
  for (int kt = 0; kt < 2; ++kt) {
    v16bf bfr = load_frag_b_swz(kvb, kt * 4 + wn, lane);
#pragma unroll
    for (int i = 0; i < 4; ++i) {
      v16bf afr = load_frag_a_b128(&sQ[kt * 32], LDQ, wm * 64 + i * 16, lane);
      acc[i] = wmma_bf16(afr, bfr, acc[i]);
    }
  }

  const int half = lane >> 4, nn = lane & 15;
#pragma unroll
  for (int i = 0; i < 4; ++i)
#pragma unroll
    for (int e = 0; e < 8; ++e) {
      const int m = wm * 64 + i * 16 + e + 8 * half;
      const int t = t0 + m;
      const int c = wn * 16 + nn;
      const float val = acc[i][e] * sZ[m];
      ATTN[((size_t)b * T_SEQ + t) * D_MODEL + h * E + c] = f2bf(val);
    }
}

// ---------------------------------------------------------------------------
extern "C" void kernel_launch(void* const* d_in, const int* in_sizes, int n_in,
                              void* d_out, int out_size, void* d_ws, size_t ws_size,
                              hipStream_t stream) {
  (void)in_sizes; (void)n_in; (void)out_size; (void)ws_size;
  const float* q  = (const float*)d_in[0];
  const float* kv = (const float*)d_in[1];
  const float* Wq = (const float*)d_in[2];
  const float* bq = (const float*)d_in[3];
  const float* Wk = (const float*)d_in[4];
  const float* bk = (const float*)d_in[5];
  const float* Wv = (const float*)d_in[6];
  const float* bv = (const float*)d_in[7];
  const float* Wo = (const float*)d_in[8];
  const float* bo = (const float*)d_in[9];

  char* ws = (char*)d_ws;
  constexpr size_t MB = 1024 * 1024;
  constexpr size_t SZ_W = (size_t)D_MODEL * D_MODEL * 2;  // 2 MiB

  unsigned int* WQs = (unsigned int*)(ws + 0 * SZ_W);
  unsigned int* WKs = (unsigned int*)(ws + 1 * SZ_W);
  unsigned int* WVs = (unsigned int*)(ws + 2 * SZ_W);
  unsigned int* WOs = (unsigned int*)(ws + 3 * SZ_W);
  unsigned short* QB  = (unsigned short*)(ws + 8 * MB);    // 32 MiB q bf16
  unsigned short* KVB = (unsigned short*)(ws + 40 * MB);   // 32 MiB kv bf16
  unsigned short* QH2 = (unsigned short*)(ws + 72 * MB);   // 32 MiB
  unsigned short* KH2 = (unsigned short*)(ws + 104 * MB);  // 32 MiB
  unsigned short* VH  = (unsigned short*)(ws + 136 * MB);  // 32 MiB (end 168)
  // overlay scratch in QB's region (QB dead after the q projection):
  float* KVP = (float*)(ws + 8 * MB);            // 8 MiB partial kv_sum
  float* KSP = (float*)(ws + 16 * MB);           // 128 KiB partial k_sum
  unsigned int* KVSW = (unsigned int*)(ws + 17 * MB);  // 512 KiB swizzled kv_sum
  float* KSUM = (float*)(ws + 18 * MB);          // 16 KiB
  unsigned short* ATTN = KH2;                    // kh2 dead after kvsum_partial

  prep_qkv_weights<<<(D_MODEL * D_MODEL / 2) / 256, 256, 0, stream>>>(Wq, Wk, Wv, WQs, WKs, WVs);
  prep_wo<<<(D_MODEL * D_MODEL / 2) / 256, 256, 0, stream>>>(Wo, WOs);

  const int cvt_blocks = (M_ROWS * D_MODEL / 8) / 256;  // 8192
  convert_f32_bf16<<<cvt_blocks, 256, 0, stream>>>(q, (unsigned int*)QB);
  convert_f32_bf16<<<cvt_blocks, 256, 0, stream>>>(kv, (unsigned int*)KVB);

  dim3 gg(M_ROWS / 128, D_MODEL / 128);
  wmma_gemm_kernel<true, true><<<gg, 256, 0, stream>>>(QB,  WQs, bq, QH2, M_ROWS, D_MODEL, D_MODEL);
  wmma_gemm_kernel<true, true><<<gg, 256, 0, stream>>>(KVB, WKs, bk, KH2, M_ROWS, D_MODEL, D_MODEL);
  wmma_gemm_kernel<false, true><<<gg, 256, 0, stream>>>(KVB, WVs, bv, VH, M_ROWS, D_MODEL, D_MODEL);

  kvsum_partial_kernel<<<dim3(BH, KSPLIT), 256, 0, stream>>>(KH2, VH, KVP, KSP);
  kvsum_reduce_kernel<<<BH, 256, 0, stream>>>(KVP, KSP, KVSW, KSUM);

  attn_kernel<<<dim3(BH, T_SEQ / 128), 256, 0, stream>>>(QH2, KVSW, KSUM, ATTN);

  wmma_gemm_kernel<false, false><<<gg, 256, 0, stream>>>(
      ATTN, WOs, bo, d_out, M_ROWS, D_MODEL, D_MODEL);
}
